// ResBlock_46480136077729
// MI455X (gfx1250) — matrix operand order repaired and checked by reading, not compiled, not run on hardware
//
#include <hip/hip_runtime.h>
#include <hip/hip_bf16.h>

// ---------------------------------------------------------------------------
// CDNA5 (gfx1250) sparse ResBlock: gather-GEMM via v_wmma_f32_16x16x32_bf16,
// LDS staging via GLOBAL_LOAD_ASYNC_TO_LDS_B128 (ASYNCcnt DMA path).
// ~88 GFLOP matmul vs ~300MB @ 23.3TB/s -> matrix-core bound; bf16 A/B,
// f32 accumulate, wave32 fragment layouts per ISA 7.12.2.
// ---------------------------------------------------------------------------

typedef __attribute__((ext_vector_type(16))) __bf16 v16bf;
typedef __attribute__((ext_vector_type(8)))  float  v8f;

#define DD   480
#define HHT  360
#define WWD  32
#define DO_  240
#define HO_  180
#define WO_  16
#define COUTC 64
#define TABN  (DD * HHT * WWD)            /* 5,529,600 (B=1) */
#define POOLN (DO_ * HO_ * WO_ * COUTC)   /* 44,236,800      */

__device__ __forceinline__ unsigned short f2bf(float f) {
    unsigned int u = __float_as_uint(f);
    u += 0x7FFFu + ((u >> 16) & 1u);     // round-to-nearest-even
    return (unsigned short)(u >> 16);
}
__device__ __forceinline__ float bf2f(unsigned short h) {
    return __uint_as_float(((unsigned int)h) << 16);
}

// Async DMA: 16 bytes global -> LDS per lane, tracked by ASYNCcnt.
// GVS mode: mem = SGPR_base + VGPR_i32 + inst_offset; VDST VGPR = LDS offset.
__device__ __forceinline__ void async_ld_b128(unsigned lds_off, unsigned gvoff,
                                              const void* base) {
    asm volatile("global_load_async_to_lds_b128 %0, %1, %2"
                 :: "v"(lds_off), "v"(gvoff), "s"(base) : "memory");
}
__device__ __forceinline__ void wait_async0() {
    asm volatile("s_wait_asynccnt 0x0" ::: "memory");
}

// ----------------------------- utility kernels -----------------------------

__global__ void fill_i32_kernel(int* __restrict__ p, int v, int n) {
    int i = blockIdx.x * blockDim.x + threadIdx.x;
    if (i < n) p[i] = v;
}

__global__ void zero_f32_kernel(float* __restrict__ p, int n) {
    int i = blockIdx.x * blockDim.x + threadIdx.x;
    if (i < n) p[i] = 0.0f;
}

__global__ void cvt_bf16_kernel(const float* __restrict__ in,
                                unsigned short* __restrict__ out, int n) {
    int i = blockIdx.x * blockDim.x + threadIdx.x;
    if (i < n) out[i] = f2bf(in[i]);
}

// fp32 [batch][K][64] -> bf16 transposed [batch][64][K] (so per-block weight
// staging becomes a contiguous async B128 copy).
__global__ void cvtT_bf16_kernel(const float* __restrict__ src,
                                 unsigned short* __restrict__ dst,
                                 int K, int total) {
    int i = blockIdx.x * blockDim.x + threadIdx.x;
    if (i >= total) return;
    int b   = i / (64 * K);
    int r   = i - b * (64 * K);
    int col = r / K;
    int j   = r - col * K;
    dst[i] = f2bf(src[(b * K + j) * 64 + col]);
}

__global__ void build_table_kernel(const int* __restrict__ coords,
                                   int* __restrict__ table, int n) {
    int i = blockIdx.x * blockDim.x + threadIdx.x;
    if (i >= n) return;
    int4 c = ((const int4*)coords)[i];  // (b,z,y,x)
    int lin = ((c.x * DD + c.y) * HHT + c.z) * WWD + c.w;
    table[lin] = i;
}

// ------------------------- submanifold conv (WMMA) -------------------------
// Block: 256 threads = 8 waves, 32 points. Wave w -> row-tile w>>2, col-tile w&3.
// LDS: nidx[9][32] | A[32][9*CIN+8] bf16 (async-gathered) | Bt[64][9*CIN+8]
// (async copy of pre-transposed weights). out = leaky(sum_k gather_k @ W[k]).

template <int CIN>
__global__ void subm_kernel(const unsigned short* __restrict__ inb,   // N x CIN bf16
                            const unsigned short* __restrict__ wtb,   // 64 x 9*CIN bf16 (W^T)
                            const int* __restrict__ table,
                            const int* __restrict__ coords,
                            unsigned short* __restrict__ outb,        // N x 64 bf16
                            const unsigned short* __restrict__ residb,// N x 64 bf16 or null
                            float* __restrict__ outf,                 // N x 64 f32 or null
                            int n, int use31) {
    constexpr int ROWS   = 32;
    constexpr int KTOT   = 9 * CIN;
    constexpr int KTOT_P = KTOT + 8;     // +16B: keeps 16B rows, banks spread

    extern __shared__ char smem[];
    int*            nidx = (int*)smem;                          // 9*ROWS ints
    unsigned short* A    = (unsigned short*)(smem + 9 * ROWS * 4);
    unsigned short* Bt   = A + ROWS * KTOT_P;                   // 64 x KTOT_P

    const unsigned loffA = (unsigned)(uintptr_t)(void*)A;   // LDS byte offsets
    const unsigned loffB = (unsigned)(uintptr_t)(void*)Bt;

    const int tid     = threadIdx.x;
    const int rowbase = blockIdx.x * ROWS;

    // Phase 1: neighbor index table lookups (9 taps x 32 rows)
    for (int e = tid; e < 9 * ROWS; e += 256) {
        int r = e & (ROWS - 1);
        int k = e >> 5;
        int p = rowbase + r;
        int idx = -1;
        if (p < n) {
            int4 c = ((const int4*)coords)[p];
            int a  = k / 3 - 1, b = k % 3 - 1;
            int dz = use31 ? a : 0;
            int dy = use31 ? 0 : a;
            int nz = c.y + dz, ny = c.z + dy, nx = c.w + b;
            if (nz >= 0 && nz < DD && ny >= 0 && ny < HHT && nx >= 0 && nx < WWD) {
                int lin = ((c.x * DD + nz) * HHT + ny) * WWD + nx;
                idx = table[lin];
            }
        }
        nidx[k * ROWS + r] = idx;
    }
    __syncthreads();

    // Phase 2: async-DMA gather of A tile (16B chunks; zero-fill invalid taps)
    {
        constexpr int CH = CIN / 8;          // 16B chunks per (row,tap) segment
        for (int e = tid; e < ROWS * 9 * CH; e += 256) {
            int r  = e / (9 * CH);
            int t  = e - r * (9 * CH);
            int k  = t / CH;
            int ch = t - k * CH;
            int nb = nidx[k * ROWS + r];
            unsigned ldst = loffA + (unsigned)((r * KTOT_P + k * CIN) * 2 + ch * 16);
            if (nb >= 0) {
                async_ld_b128(ldst, (unsigned)(nb * CIN * 2 + ch * 16), inb);
            } else {
                *(uint4*)((char*)A + (ldst - loffA)) = make_uint4(0u, 0u, 0u, 0u);
            }
        }
    }

    // Phase 3: async-DMA copy of transposed weights (contiguous in global)
    for (int e = tid; e < 64 * (KTOT / 8); e += 256) {
        int col = e / (KTOT / 8);
        int ch  = e - col * (KTOT / 8);
        unsigned ldst = loffB + (unsigned)(col * KTOT_P * 2 + ch * 16);
        async_ld_b128(ldst, (unsigned)(e * 16), wtb);
    }
    wait_async0();
    __syncthreads();

    // Phase 4: WMMA accumulation over K chunks of 32
    const int wave = tid >> 5, lane = tid & 31;
    const int rowt = wave >> 2, colt = wave & 3;
    const int half = lane >> 4, ln = lane & 15;
    const unsigned short* Arow = A  + (rowt * 16 + ln) * KTOT_P;
    const unsigned short* Bcol = Bt + (colt * 16 + ln) * KTOT_P;

    v8f c = {0.f, 0.f, 0.f, 0.f, 0.f, 0.f, 0.f, 0.f};
    constexpr int NKK = KTOT / 32;
    for (int kk = 0; kk < NKK; ++kk) {
        union { unsigned int u[8]; v16bf v; } af, bfr;
        int kbA = kk * 32 + 8 * half;
#pragma unroll
        for (int v = 0; v < 8; ++v) {    // A: V0-3 -> K 0..7, V4-7 -> K 16..23 (+8*half)
            int ka = kbA + (v < 4 ? 2 * v : 8 + 2 * v);
            af.u[v] = *(const unsigned int*)&Arow[ka];
        }
        int kbB = kk * 32 + 8 * half;
#pragma unroll
        for (int v = 0; v < 8; ++v)      // B: V0-3 -> K 0..7, V4-7 -> K 16..23 (+8*half)
            bfr.u[v] = *(const unsigned int*)&Bcol[kbB + (v < 4 ? 2 * v : 8 + 2 * v)];
        c = __builtin_amdgcn_wmma_f32_16x16x32_bf16(false, af.v, false, bfr.v,
                                                    (short)0, c, false, false);
    }

    // Phase 5: leaky + residual + store (C layout: M = r + 8*half, N = ln)
#pragma unroll
    for (int r = 0; r < 8; ++r) {
        int row = rowbase + rowt * 16 + r + 8 * half;
        if (row < n) {
            int col = colt * 16 + ln;
            float v = c[r];
            v = v > 0.f ? v : 0.01f * v;
            if (residb) v += bf2f(residb[row * 64 + col]);
            outb[row * 64 + col] = f2bf(v);
            if (outf) outf[row * 64 + col] = v;
        }
    }
}

// ------------------------------ strided pool -------------------------------
// grid (ceil(N/32), 27): block = (32-point tile, tap k). Early-out when no
// point maps to a valid output cell. 2 WMMAs/wave then atomic f32 scatter.

__global__ void pool_kernel(const unsigned short* __restrict__ rAb, // N x 64 bf16
                            const unsigned short* __restrict__ wptb,// 27 x 64 x 64 bf16 (W^T)
                            const int* __restrict__ coords,
                            float* __restrict__ pooled, int n) {
    constexpr int KP = 72;  // 64 + 8: 16B rows + bank spread
    __shared__ __align__(16) int olin[32];
    __shared__ int anyv;
    __shared__ __align__(16) unsigned short As[32 * KP];
    __shared__ __align__(16) unsigned short Bts[64 * KP];

    const int tid = threadIdx.x;
    const int rowbase = blockIdx.x * 32;
    const int k  = blockIdx.y;
    const int kd = k / 9, kh = (k / 3) % 3, kw = k % 3;

    if (tid == 0) anyv = 0;
    __syncthreads();

    if (tid < 32) {
        int p = rowbase + tid, o = -1;
        if (p < n) {
            int4 c = ((const int4*)coords)[p];
            int oz = c.y + 1 - kd, oy = c.z + 1 - kh, ox = c.w + 1 - kw;
            if (!((oz & 1) | (oy & 1) | (ox & 1))) {   // stride-2 parity
                oz >>= 1; oy >>= 1; ox >>= 1;
                if (oz >= 0 && oz < DO_ && oy >= 0 && oy < HO_ &&
                    ox >= 0 && ox < WO_) {
                    o = ((c.x * DO_ + oz) * HO_ + oy) * WO_ + ox;
                    anyv = 1;
                }
            }
        }
        olin[tid] = o;
    }
    __syncthreads();
    if (!anyv) return;

    // Async-DMA stage: A tile rows (128B each) + tap weights (contiguous 8KB)
    const unsigned loffA = (unsigned)(uintptr_t)(void*)As;
    const unsigned loffB = (unsigned)(uintptr_t)(void*)Bts;
    for (int e = tid; e < 32 * 8; e += 256) {
        int r = e >> 3, ch = e & 7;
        int p = rowbase + r;
        unsigned ldst = loffA + (unsigned)(r * KP * 2 + ch * 16);
        if (p < n) {
            async_ld_b128(ldst, (unsigned)(p * 128 + ch * 16), rAb);
        } else {
            *(uint4*)((char*)As + (ldst - loffA)) = make_uint4(0u, 0u, 0u, 0u);
        }
    }
    for (int e = tid; e < 64 * 8; e += 256) {
        int col = e >> 3, ch = e & 7;
        unsigned ldst = loffB + (unsigned)(col * KP * 2 + ch * 16);
        async_ld_b128(ldst, (unsigned)(k * 8192 + e * 16), wptb);
    }
    wait_async0();
    __syncthreads();

    const int wave = tid >> 5, lane = tid & 31;
    const int rowt = wave >> 2, colt = wave & 3;
    const int half = lane >> 4, ln = lane & 15;
    const unsigned short* Arow = As  + (rowt * 16 + ln) * KP;
    const unsigned short* Bcol = Bts + (colt * 16 + ln) * KP;

    v8f c = {0.f, 0.f, 0.f, 0.f, 0.f, 0.f, 0.f, 0.f};
#pragma unroll
    for (int kk = 0; kk < 2; ++kk) {
        union { unsigned int u[8]; v16bf v; } af, bfr;
        int kbA = kk * 32 + 8 * half;
#pragma unroll
        for (int v = 0; v < 8; ++v) {
            int ka = kbA + (v < 4 ? 2 * v : 8 + 2 * v);
            af.u[v] = *(const unsigned int*)&Arow[ka];
        }
        int kbB = kk * 32 + 8 * half;
#pragma unroll
        for (int v = 0; v < 8; ++v)
            bfr.u[v] = *(const unsigned int*)&Bcol[kbB + (v < 4 ? 2 * v : 8 + 2 * v)];
        c = __builtin_amdgcn_wmma_f32_16x16x32_bf16(false, af.v, false, bfr.v,
                                                    (short)0, c, false, false);
    }

    // Preload this lane's 8 olin entries (two b128 LDS reads), then scatter.
    const int obase = rowt * 16 + 8 * half;
    int4 oa = *(const int4*)&olin[obase];
    int4 ob = *(const int4*)&olin[obase + 4];
    int oarr[8] = {oa.x, oa.y, oa.z, oa.w, ob.x, ob.y, ob.z, ob.w};
#pragma unroll
    for (int r = 0; r < 8; ++r) {
        if (oarr[r] >= 0)
            atomicAdd(&pooled[oarr[r] * 64 + colt * 16 + ln], c[r]);
    }
}

// ------------------------------- host launch -------------------------------

static inline size_t align256(size_t x) { return (x + 255) & ~(size_t)255; }

extern "C" void kernel_launch(void* const* d_in, const int* in_sizes, int n_in,
                              void* d_out, int out_size, void* d_ws, size_t ws_size,
                              hipStream_t stream) {
    const float* feats  = (const float*)d_in[0];
    const int*   coords = (const int*)d_in[1];
    const float* W1     = (const float*)d_in[2];
    const float* W1_2   = (const float*)d_in[3];
    const float* W2     = (const float*)d_in[4];
    const float* W3     = (const float*)d_in[5];
    const float* Wpool  = (const float*)d_in[6];
    const int n = in_sizes[0] / 32;               // N = 200000

    // workspace carve-out
    size_t off = 0;
    int* tab = (int*)((char*)d_ws + off);                        off = align256(off + (size_t)TABN * 4);
    unsigned short* w1t  = (unsigned short*)((char*)d_ws + off); off = align256(off + (size_t)9 * 32 * 64 * 2);
    unsigned short* w12t = (unsigned short*)((char*)d_ws + off); off = align256(off + (size_t)9 * 64 * 64 * 2);
    unsigned short* w2t  = (unsigned short*)((char*)d_ws + off); off = align256(off + (size_t)9 * 32 * 64 * 2);
    unsigned short* w3t  = (unsigned short*)((char*)d_ws + off); off = align256(off + (size_t)9 * 64 * 64 * 2);
    unsigned short* wpt  = (unsigned short*)((char*)d_ws + off); off = align256(off + (size_t)27 * 64 * 64 * 2);
    unsigned short* featsb = (unsigned short*)((char*)d_ws + off); off = align256(off + (size_t)n * 32 * 2);
    unsigned short* bufA = (unsigned short*)((char*)d_ws + off); off = align256(off + (size_t)n * 64 * 2);
    unsigned short* bufB = (unsigned short*)((char*)d_ws + off); off = align256(off + (size_t)n * 64 * 2);
    unsigned short* bufC = (unsigned short*)((char*)d_ws + off); off = align256(off + (size_t)n * 64 * 2);

    float* pooled = (float*)d_out;
    float* rA_out = (float*)d_out + (size_t)POOLN;

    // 1) hash table
    fill_i32_kernel<<<(TABN + 255) / 256, 256, 0, stream>>>(tab, -1, TABN);
    build_table_kernel<<<(n + 255) / 256, 256, 0, stream>>>(coords, tab, n);

    // 2) conversions: weights transposed ([64][K] bf16) so staging is a
    //    straight async B128 copy; features plain bf16.
    cvtT_bf16_kernel<<<(9*32*64 + 255)/256, 256, 0, stream>>>(W1,   w1t,  9*32, 9*32*64);
    cvtT_bf16_kernel<<<(9*64*64 + 255)/256, 256, 0, stream>>>(W1_2, w12t, 9*64, 9*64*64);
    cvtT_bf16_kernel<<<(9*32*64 + 255)/256, 256, 0, stream>>>(W2,   w2t,  9*32, 9*32*64);
    cvtT_bf16_kernel<<<(9*64*64 + 255)/256, 256, 0, stream>>>(W3,   w3t,  9*64, 9*64*64);
    cvtT_bf16_kernel<<<(27*64*64 + 255)/256, 256, 0, stream>>>(Wpool, wpt, 64,  27*64*64);
    cvt_bf16_kernel<<<(n * 32 + 255) / 256, 256, 0, stream>>>(feats, featsb, n * 32);

    const int nblk = (n + 31) / 32;
    const size_t lds32 = (size_t)(9 * 32 * 4) + (size_t)(32 + 64) * (9 * 32 + 8) * 2; //  ~58 KB
    const size_t lds64 = (size_t)(9 * 32 * 4) + (size_t)(32 + 64) * (9 * 64 + 8) * 2; // ~113 KB (<320 KB WGP)

    // 3) two-branch submanifold convs
    subm_kernel<32><<<nblk, 256, lds32, stream>>>(featsb, w1t, tab, coords, bufA, nullptr, nullptr, n, 1);
    subm_kernel<64><<<nblk, 256, lds64, stream>>>(bufA, w12t, tab, coords, bufB, nullptr, nullptr, n, 0);
    subm_kernel<32><<<nblk, 256, lds32, stream>>>(featsb, w2t, tab, coords, bufA, nullptr, nullptr, n, 0);
    subm_kernel<64><<<nblk, 256, lds64, stream>>>(bufA, w3t, tab, coords, bufC, bufB, rA_out, n, 1);

    // 4) strided sparse->dense pool (zero then atomic scatter)
    zero_f32_kernel<<<(POOLN + 255) / 256, 256, 0, stream>>>(pooled, POOLN);
    dim3 pgrid(nblk, 27);
    pool_kernel<<<pgrid, 256, 0, stream>>>(bufC, wpt, coords, pooled, n);
}
